// CausalSelfAttention_57346403336555
// MI455X (gfx1250) — compile-verified, matrix operand order checked by reading, not run on hardware
//
#include <hip/hip_runtime.h>

// ---------------------------------------------------------------------------
// Problem constants
// ---------------------------------------------------------------------------
#define BB 2
#define TT 2048
#define CC 768
#define HH 12
#define DD 64
#define BT (BB * TT)        // 4096
#define F3 (3 * CC)         // 2304

typedef __attribute__((ext_vector_type(16))) __bf16 v16bf;
typedef __attribute__((ext_vector_type(8)))  __bf16 v8bf;
typedef __attribute__((ext_vector_type(8)))  float  v8f;

// -ln(10000)/32
#define NEG_LN_THETA_OVER_HALFD (-0.28782313662425575f)

__device__ __forceinline__ v8f wmma_bf16(v16bf a, v16bf b, v8f c) {
  // D = A(16x32 bf16) * B(32x16 bf16) + C(16x16 f32)
  return __builtin_amdgcn_wmma_f32_16x16x32_bf16(false, a, false, b,
                                                 (short)0, c, false, false);
}

__device__ __forceinline__ v8f zero8() {
  v8f z;
#pragma unroll
  for (int i = 0; i < 8; ++i) z[i] = 0.0f;
  return z;
}

// Load a 16x32 bf16 fragment from row-major storage (row stride ld elements).
// ISA 16-bit A layout: lane&15 = row, (lane>>4) selects K-chunks
// {0..7,16..23} vs {8..15,24..31}. Same layout used for B fed as NxK rows.
__device__ __forceinline__ v16bf load_frag(const __bf16* base, int ld, int lane) {
  const int r  = lane & 15;
  const int hh = lane >> 4;
  const __bf16* p = base + (size_t)r * ld + hh * 8;
  union { v16bf v; v8bf h[2]; } u;
  u.h[0] = *(const v8bf*)(p);
  u.h[1] = *(const v8bf*)(p + 16);
  return u.v;
}

// ---------------------------------------------------------------------------
// Kernel 0: fp32 -> bf16 conversion
// ---------------------------------------------------------------------------
__global__ void cvt_bf16_kernel(const float* __restrict__ in,
                                __bf16* __restrict__ out, int n) {
  int i = blockIdx.x * blockDim.x + threadIdx.x;
  if (i < n) out[i] = (__bf16)in[i];
}

// ---------------------------------------------------------------------------
// Kernel 1: QKV GEMM (BT x F3 x CC) + fused RoPE + layout shuffle.
//   qr, kr : (B,H,T,D) bf16 (RoPE applied)
//   vT     : (B,H,D,T) bf16 (transposed so PV B-matrix is K-contiguous)
// One wave -> 16x64 output tile (head-aligned for q/k regions).
// ---------------------------------------------------------------------------
__global__ __launch_bounds__(128) void qkv_rope_kernel(
    const __bf16* __restrict__ xb,   // (BT, CC) row-major
    const __bf16* __restrict__ wb,   // (F3, CC) row-major
    __bf16* __restrict__ qr, __bf16* __restrict__ kr, __bf16* __restrict__ vT) {
  const int lane = threadIdx.x & 31;
  const int gw   = blockIdx.x * 4 + (threadIdx.x >> 5);  // 0 .. 9215
  const int mt   = gw & 255;   // 256 M-tiles of 16
  const int nt   = gw >> 8;    // 36  N-tiles of 64

  v8f acc[4];
#pragma unroll
  for (int i = 0; i < 4; ++i) acc[i] = zero8();

  const __bf16* arow = xb + (size_t)(mt * 16) * CC;
  const __bf16* brow = wb + (size_t)(nt * 64) * CC;
  for (int kk = 0; kk < CC; kk += 32) {
    // hint next A-slice into cache (global_prefetch_b8)
    if (kk + 32 < CC) __builtin_prefetch(arow + kk + 32 + (size_t)(lane & 15) * CC, 0, 1);
    v16bf a = load_frag(arow + kk, CC, lane);
#pragma unroll
    for (int i = 0; i < 4; ++i) {
      v16bf b = load_frag(brow + (size_t)(i * 16) * CC + kk, CC, lane);
      acc[i] = wmma_bf16(a, b, acc[i]);
    }
  }

  // Epilogue: C layout -> row = rowBase+v, col = colN (+ subtile offset)
  const int colN    = lane & 15;
  const int rowBase = (lane >> 4) * 8;
  const int fb      = nt * 64;
#pragma unroll
  for (int i = 0; i < 4; ++i) {
    const int f0   = fb + i * 16;       // wave-uniform
    const int reg3 = f0 / CC;           // 0=q, 1=k, 2=v (uniform)
    const int fr0  = f0 - reg3 * CC;
    const int h    = fr0 >> 6;          // head (uniform)
    const int d    = (fr0 & 63) + colN; // 0..63 within head
    // inv_freq = theta^(-(d%32)/32) = exp((d%32) * -ln(theta)/32)
    const float invf = __expf((float)(d & 31) * NEG_LN_THETA_OVER_HALFD);
#pragma unroll
    for (int v = 0; v < 8; ++v) {
      const int m = mt * 16 + rowBase + v;
      const int b = m >> 11;            // m / TT
      const int t = m & (TT - 1);
      float val = acc[i][v];
      if (reg3 == 2) {
        vT[((size_t)(b * HH + h) * DD + d) * TT + t] = (__bf16)val;
      } else {
        const float pairv = acc[i ^ 2][v];           // element at d +/- 32
        const float rot   = (d < 32) ? -pairv : pairv;
        const float ang   = (float)t * invf;
        const float o     = val * __cosf(ang) + rot * __sinf(ang);
        __bf16* dst = reg3 ? kr : qr;
        dst[((size_t)(b * HH + h) * TT + t) * DD + d] = (__bf16)o;
      }
    }
  }
}

// ---------------------------------------------------------------------------
// Kernel 2: causal flash attention. One wave per 16-query tile.
//   S(16x32) via 4 WMMAs, online softmax in C layout, P transposed to
//   A layout through LDS, PV(16x64) via 4 WMMAs per 32 KV columns.
// ---------------------------------------------------------------------------
__global__ __launch_bounds__(128) void attn_kernel(
    const __bf16* __restrict__ qr, const __bf16* __restrict__ kr,
    const __bf16* __restrict__ vT, __bf16* __restrict__ att) {
  __shared__ __align__(16) __bf16 Pl[4][16 * 32];

  const int wave = threadIdx.x >> 5;
  const int lane = threadIdx.x & 31;
  const int gw   = blockIdx.x * 4 + wave;   // 0 .. 3071
  const int qt   = gw & 127;                // T/16 = 128 q-tiles
  const int bh   = gw >> 7;                 // 0 .. 23
  const int qb   = qt * 16;

  const __bf16* qp = qr + (size_t)bh * TT * DD;
  const __bf16* kp = kr + (size_t)bh * TT * DD;
  const __bf16* vp = vT + (size_t)bh * DD * TT;

  const v16bf qa0 = load_frag(qp + (size_t)qb * DD +  0, DD, lane);
  const v16bf qa1 = load_frag(qp + (size_t)qb * DD + 32, DD, lane);

  v8f o[4];
#pragma unroll
  for (int n = 0; n < 4; ++n) o[n] = zero8();
  float mrow[8], lrow[8];
#pragma unroll
  for (int v = 0; v < 8; ++v) { mrow[v] = -3.0e38f; lrow[v] = 0.0f; }

  const int colN    = lane & 15;
  const int rowBase = (lane >> 4) * 8;
  const int numJ    = (qb + 47) >> 5;       // KV tiles of 32, covers cols <= qb+15

  for (int jt = 0; jt < numJ; ++jt) {
    const int j0 = jt * 32;

    // ---- S = q * k^T (16x32), scaled + causally masked -------------------
    v8f s[2];
#pragma unroll
    for (int sub = 0; sub < 2; ++sub) {
      const __bf16* kb = kp + (size_t)(j0 + sub * 16) * DD;
      v16bf b0 = load_frag(kb,      DD, lane);
      v16bf b1 = load_frag(kb + 32, DD, lane);
      s[sub] = wmma_bf16(qa1, b1, wmma_bf16(qa0, b0, zero8()));
    }
#pragma unroll
    for (int sub = 0; sub < 2; ++sub)
#pragma unroll
      for (int v = 0; v < 8; ++v) {
        const int row = qb + rowBase + v;
        const int col = j0 + sub * 16 + colN;
        const float sv = s[sub][v] * 0.125f;         // 1/sqrt(64)
        s[sub][v] = (col <= row) ? sv : -3.0e38f;
      }

    // ---- online softmax (rows live across each 16-lane half) -------------
#pragma unroll
    for (int v = 0; v < 8; ++v) {
      float rm = fmaxf(s[0][v], s[1][v]);
#pragma unroll
      for (int msk = 1; msk < 16; msk <<= 1) rm = fmaxf(rm, __shfl_xor(rm, msk, 32));
      const float mn  = fmaxf(mrow[v], rm);
      const float fsc = __expf(mrow[v] - mn);
      const float p0  = __expf(s[0][v] - mn);
      const float p1  = __expf(s[1][v] - mn);
      s[0][v] = p0; s[1][v] = p1;
      float rs = p0 + p1;
#pragma unroll
      for (int msk = 1; msk < 16; msk <<= 1) rs += __shfl_xor(rs, msk, 32);
      lrow[v] = lrow[v] * fsc + rs;
      mrow[v] = mn;
#pragma unroll
      for (int n = 0; n < 4; ++n) o[n][v] *= fsc;
    }

    // ---- P: C layout -> A layout via LDS ---------------------------------
    __bf16* pl = Pl[wave];
#pragma unroll
    for (int sub = 0; sub < 2; ++sub)
#pragma unroll
      for (int v = 0; v < 8; ++v)
        pl[(rowBase + v) * 32 + sub * 16 + colN] = (__bf16)s[sub][v];
    asm volatile("s_wait_dscnt 0x0" ::: "memory");   // wave-local LDS RAW

    v16bf pa;
    {
      const __bf16* pp = pl + (lane & 15) * 32 + (lane >> 4) * 8;
      union { v16bf v; v8bf h[2]; } u;
      u.h[0] = *(const v8bf*)(pp);
      u.h[1] = *(const v8bf*)(pp + 16);
      pa = u.v;
    }

    // ---- O += P * V (V^T stored (D,T): K-contiguous B fragments) ---------
#pragma unroll
    for (int n = 0; n < 4; ++n) {
      v16bf bv = load_frag(vp + (size_t)(n * 16) * TT + j0, TT, lane);
      o[n] = wmma_bf16(pa, bv, o[n]);
    }
  }

  // ---- normalize + write (B,T,C) bf16 ------------------------------------
  const int b = bh / HH, h = bh % HH;
#pragma unroll
  for (int n = 0; n < 4; ++n)
#pragma unroll
    for (int v = 0; v < 8; ++v) {
      const int t = qb + rowBase + v;
      const float val = o[n][v] / lrow[v];
      att[((size_t)(b * TT + t)) * CC + h * DD + n * 16 + colN] = (__bf16)val;
    }
}

// ---------------------------------------------------------------------------
// Kernel 3: output projection (BT x CC x CC), fp32 output.
// ---------------------------------------------------------------------------
__global__ __launch_bounds__(128) void proj_kernel(
    const __bf16* __restrict__ att,  // (BT, CC) bf16
    const __bf16* __restrict__ wb,   // (CC, CC) bf16
    float* __restrict__ out) {       // (BT, CC) fp32
  const int lane = threadIdx.x & 31;
  const int gw   = blockIdx.x * 4 + (threadIdx.x >> 5);  // 0 .. 3071
  const int mt   = gw & 255;
  const int nt   = gw >> 8;    // 12 N-tiles of 64

  v8f acc[4];
#pragma unroll
  for (int i = 0; i < 4; ++i) acc[i] = zero8();

  const __bf16* arow = att + (size_t)(mt * 16) * CC;
  const __bf16* brow = wb  + (size_t)(nt * 64) * CC;
  for (int kk = 0; kk < CC; kk += 32) {
    if (kk + 32 < CC) __builtin_prefetch(arow + kk + 32 + (size_t)(lane & 15) * CC, 0, 1);
    v16bf a = load_frag(arow + kk, CC, lane);
#pragma unroll
    for (int i = 0; i < 4; ++i) {
      v16bf b = load_frag(brow + (size_t)(i * 16) * CC + kk, CC, lane);
      acc[i] = wmma_bf16(a, b, acc[i]);
    }
  }

  const int colN    = lane & 15;
  const int rowBase = (lane >> 4) * 8;
#pragma unroll
  for (int i = 0; i < 4; ++i)
#pragma unroll
    for (int v = 0; v < 8; ++v) {
      const int m = mt * 16 + rowBase + v;
      out[(size_t)m * CC + nt * 64 + i * 16 + colN] = acc[i][v];
    }
}

// ---------------------------------------------------------------------------
// Host-side launch
// ---------------------------------------------------------------------------
extern "C" void kernel_launch(void* const* d_in, const int* in_sizes, int n_in,
                              void* d_out, int out_size, void* d_ws, size_t ws_size,
                              hipStream_t stream) {
  const float* x     = (const float*)d_in[0];   // (B,T,C)
  const float* wqkv  = (const float*)d_in[1];   // (3C,C)
  const float* wproj = (const float*)d_in[2];   // (C,C)
  float* out = (float*)d_out;

  // Workspace carve-up (bf16 buffers), 256B aligned.
  const size_t N_X   = (size_t)BT * CC;        // 3,145,728
  const size_t N_WQ  = (size_t)F3 * CC;        // 1,769,472
  const size_t N_WP  = (size_t)CC * CC;        //   589,824
  const size_t N_HTD = (size_t)BB * HH * TT * DD;  // 3,145,728

  char* p = (char*)d_ws;
  auto carve = [&](size_t nelem) {
    __bf16* q = (__bf16*)p;
    size_t bytes = (nelem * sizeof(__bf16) + 255) & ~(size_t)255;
    p += bytes;
    return q;
  };
  __bf16* xb  = carve(N_X);
  __bf16* wqb = carve(N_WQ);
  __bf16* wpb = carve(N_WP);
  __bf16* qr  = carve(N_HTD);
  __bf16* kr  = carve(N_HTD);
  __bf16* vT  = carve(N_HTD);
  __bf16* att = carve(N_X);

  // 0) fp32 -> bf16
  cvt_bf16_kernel<<<(int)((N_X  + 255) / 256), 256, 0, stream>>>(x,     xb,  (int)N_X);
  cvt_bf16_kernel<<<(int)((N_WQ + 255) / 256), 256, 0, stream>>>(wqkv,  wqb, (int)N_WQ);
  cvt_bf16_kernel<<<(int)((N_WP + 255) / 256), 256, 0, stream>>>(wproj, wpb, (int)N_WP);

  // 1) QKV GEMM + RoPE + transpose-V  (9216 waves, 4 per block)
  qkv_rope_kernel<<<2304, 128, 0, stream>>>(xb, wqb, qr, kr, vT);

  // 2) causal flash attention (3072 waves, 4 per block)
  attn_kernel<<<768, 128, 0, stream>>>(qr, kr, vT, att);

  // 3) output projection (3072 waves, 4 per block)
  proj_kernel<<<768, 128, 0, stream>>>(att, wpb, out);
}
